// GATBlock_71116068487800
// MI455X (gfx1250) — matrix operand order repaired and checked by reading, not compiled, not run on hardware
//
#include <hip/hip_runtime.h>

// ---------------------------------------------------------------------------
// 3-layer GAT for MI455X (gfx1250, wave32).
// GEMMs via v_wmma_f32_16x16x32_bf16 (bf16 operands, f32 accumulate),
// software-pipelined K loop so loads overlap matrix ops.
// Attention softmax/aggregation in f32 VALU (contiguous 16-edge segments).
// ---------------------------------------------------------------------------

#define NNODES 16384
#define DEG    16

typedef __attribute__((ext_vector_type(16))) __bf16       v16bf;
typedef __attribute__((ext_vector_type(8)))  float        v8f;
typedef __attribute__((ext_vector_type(4)))  unsigned int v4u;

__device__ __forceinline__ unsigned short f32_to_bf16_bits(float f) {
  // round-to-nearest-even f32 -> bf16
  unsigned int u = __builtin_bit_cast(unsigned int, f);
  unsigned int lsb = (u >> 16) & 1u;
  u += 0x7fffu + lsb;
  return (unsigned short)(u >> 16);
}

union FragU { v16bf v; v4u q[2]; };

// load a 16-element bf16 fragment as two b128 chunks `gap` elements apart
__device__ __forceinline__ v16bf load_frag(const unsigned short* p, int gap) {
  FragU f;
  f.q[0] = *(const v4u*)(p);
  f.q[1] = *(const v4u*)(p + gap);
  return f.v;
}

// --------------------------- format conversion -----------------------------

__global__ void __launch_bounds__(256)
cvt_bf16_kernel(const float* __restrict__ in, unsigned short* __restrict__ out, int n) {
  for (int i = blockIdx.x * blockDim.x + threadIdx.x; i < n; i += gridDim.x * blockDim.x)
    out[i] = f32_to_bf16_bits(in[i]);
}

// W [Fin x Fout] f32  ->  Wt [Fout x Fin] bf16  (so B fragments load contiguous K)
__global__ void __launch_bounds__(256)
transpose_bf16_kernel(const float* __restrict__ W, unsigned short* __restrict__ Wt,
                      int Fin, int Fout) {
  int t = blockIdx.x * blockDim.x + threadIdx.x;
  if (t >= Fin * Fout) return;
  int fi = t % Fin;
  int fo = t / Fin;
  Wt[(size_t)fo * Fin + fi] = f32_to_bf16_bits(W[(size_t)fi * Fout + fo]);
}

// ------------------------------ WMMA GEMM ----------------------------------
// Hout[N x Fout] (f32) = X[N x Fin] (bf16) * Wt^T  (Wt stored [Fout x Fin] bf16)
// 8 waves / block; each wave computes a 16x64 tile (4 accumulators).
// K loop (step 32) is software-pipelined: next fragments load while the
// current four v_wmma_f32_16x16x32_bf16 issue.
__global__ void __launch_bounds__(256)
gemm_bf16_wmma_kernel(const unsigned short* __restrict__ X,
                      const unsigned short* __restrict__ Wt,
                      float* __restrict__ Hout,
                      int Fin, int Fout) {
  const int lane = threadIdx.x & 31;
  const int wave = threadIdx.x >> 5;
  const int m0   = blockIdx.x * 128 + wave * 16;   // 16 output rows per wave
  const int n0   = blockIdx.y * 64;                // 64 output cols per wave

  // A fragment (16x32 bf16): lane l<16 -> M=l, K=[0,16)
  //                          lane l>=16 -> same M, K=[16,32)
  const int am  = m0 + (lane & 15);
  const int akb = (lane >> 4) << 4;                // 0 or 16
  const unsigned short* aptr = X + (size_t)am * Fin + akb;

  // B fragment (32x16 bf16): lane l<16 -> N=l,  K=[0,16); l>=16 -> N=l-16, K=[16,32)
  const int bn  = lane & 15;
  const int bkb = (lane >> 4) << 4;                // 0 or 16
  const unsigned short* bptr0 = Wt + (size_t)(n0 +  0 + bn) * Fin + bkb;
  const unsigned short* bptr1 = Wt + (size_t)(n0 + 16 + bn) * Fin + bkb;
  const unsigned short* bptr2 = Wt + (size_t)(n0 + 32 + bn) * Fin + bkb;
  const unsigned short* bptr3 = Wt + (size_t)(n0 + 48 + bn) * Fin + bkb;

  v8f acc0 = {}, acc1 = {}, acc2 = {}, acc3 = {};

  // pipeline prologue: fragments for k0 = 0
  v16bf a_c  = load_frag(aptr,   8);
  v16bf b0_c = load_frag(bptr0,  8);
  v16bf b1_c = load_frag(bptr1,  8);
  v16bf b2_c = load_frag(bptr2,  8);
  v16bf b3_c = load_frag(bptr3,  8);

  for (int k0 = 32; k0 < Fin; k0 += 32) {
    // issue next-iteration loads first: the 4 WMMAs below have no dependence
    // on them, so loadcnt waits sink past the matrix ops.
    v16bf a_n  = load_frag(aptr  + k0,  8);
    v16bf b0_n = load_frag(bptr0 + k0,  8);
    v16bf b1_n = load_frag(bptr1 + k0,  8);
    v16bf b2_n = load_frag(bptr2 + k0,  8);
    v16bf b3_n = load_frag(bptr3 + k0,  8);
    if (k0 + 32 < Fin)                              // CDNA5 global_prefetch of A panel
      __builtin_prefetch((const void*)(aptr + k0 + 32), 0, 3);

    acc0 = __builtin_amdgcn_wmma_f32_16x16x32_bf16(false, a_c, false, b0_c, (short)0, acc0, false, false);
    acc1 = __builtin_amdgcn_wmma_f32_16x16x32_bf16(false, a_c, false, b1_c, (short)0, acc1, false, false);
    acc2 = __builtin_amdgcn_wmma_f32_16x16x32_bf16(false, a_c, false, b2_c, (short)0, acc2, false, false);
    acc3 = __builtin_amdgcn_wmma_f32_16x16x32_bf16(false, a_c, false, b3_c, (short)0, acc3, false, false);

    a_c = a_n; b0_c = b0_n; b1_c = b1_n; b2_c = b2_n; b3_c = b3_n;
  }

  // pipeline epilogue
  acc0 = __builtin_amdgcn_wmma_f32_16x16x32_bf16(false, a_c, false, b0_c, (short)0, acc0, false, false);
  acc1 = __builtin_amdgcn_wmma_f32_16x16x32_bf16(false, a_c, false, b1_c, (short)0, acc1, false, false);
  acc2 = __builtin_amdgcn_wmma_f32_16x16x32_bf16(false, a_c, false, b2_c, (short)0, acc2, false, false);
  acc3 = __builtin_amdgcn_wmma_f32_16x16x32_bf16(false, a_c, false, b3_c, (short)0, acc3, false, false);

  // C/D layout: VGPR r -> M = r + 8*(lane>=16), N = lane&15
  const int crow = m0 + ((lane >> 4) << 3);
  const int ccol = n0 + (lane & 15);
#pragma unroll
  for (int r = 0; r < 8; ++r) {
    float* orow = Hout + (size_t)(crow + r) * Fout + ccol;
    orow[0]  = acc0[r];
    orow[16] = acc1[r];
    orow[32] = acc2[r];
    orow[48] = acc3[r];
  }
}

// -------------------------- attention scores -------------------------------
// EL[n,h] = <H[n,h,:], al[h,:]> ; ER likewise.
// One wave per (node, head): lanes read contiguous d (coalesced), butterfly
// reduce across the wave.
__global__ void __launch_bounds__(256)
attn_scores_kernel(const float* __restrict__ Hbuf,
                   const float* __restrict__ al, const float* __restrict__ ar,
                   float* __restrict__ EL, float* __restrict__ ER, int Hn, int D) {
  int wid  = (blockIdx.x * blockDim.x + threadIdx.x) >> 5;
  int lane = threadIdx.x & 31;
  if (wid >= NNODES * Hn) return;
  int n = wid / Hn, h = wid % Hn;
  const float* row = Hbuf + (size_t)n * Hn * D + (size_t)h * D;
  const float* alh = al + h * D;
  const float* arh = ar + h * D;
  float sl = 0.f, sr = 0.f;
  for (int d = lane; d < D; d += 32) {
    float v = row[d];
    sl = fmaf(v, alh[d], sl);
    sr = fmaf(v, arh[d], sr);
  }
#pragma unroll
  for (int off = 16; off > 0; off >>= 1) {
    sl += __shfl_xor(sl, off, 32);
    sr += __shfl_xor(sr, off, 32);
  }
  if (lane == 0) {
    EL[wid] = sl;
    ER[wid] = sr;
  }
}

// ---------------------- softmax-aggregate (+ELU) ---------------------------
// dst segments are contiguous: edges of node n are src[16n .. 16n+15].
// One wave per (node, head); lanes cover the feature dim.
__global__ void __launch_bounds__(256)
gat_aggregate_elu_kernel(const float* __restrict__ Hbuf,
                         const int* __restrict__ src,
                         const float* __restrict__ EL, const float* __restrict__ ER,
                         const float* __restrict__ bias,
                         unsigned short* __restrict__ Xnext,   // bf16 for next GEMM
                         int Hn, int D) {
  int wid  = (blockIdx.x * blockDim.x + threadIdx.x) >> 5;
  int lane = threadIdx.x & 31;
  if (wid >= NNODES * Hn) return;
  int n = wid / Hn, h = wid % Hn;

  int s[DEG];
  float e[DEG];
  float er_n = ER[n * Hn + h];
  float m = -3.0e38f;
#pragma unroll
  for (int k = 0; k < DEG; ++k) {
    int sk = src[n * DEG + k];
    s[k] = sk;
    float ev = EL[sk * Hn + h] + er_n;
    ev = ev > 0.f ? ev : 0.2f * ev;                 // leaky_relu, slope 0.2
    e[k] = ev;
    m = fmaxf(m, ev);
  }
  float denom = 0.f;
#pragma unroll
  for (int k = 0; k < DEG; ++k) { e[k] = __expf(e[k] - m); denom += e[k]; }
  float inv = 1.f / denom;

  float acc[4] = {0.f, 0.f, 0.f, 0.f};
  for (int k = 0; k < DEG; ++k) {
    const float* hr = Hbuf + (size_t)s[k] * Hn * D + (size_t)h * D;
    float w = e[k] * inv;
#pragma unroll
    for (int i = 0; i < 4; ++i) {
      int d = lane + 32 * i;
      if (d < D) acc[i] = fmaf(w, hr[d], acc[i]);
    }
  }
  unsigned short* orow = Xnext + (size_t)n * Hn * D + (size_t)h * D;
#pragma unroll
  for (int i = 0; i < 4; ++i) {
    int d = lane + 32 * i;
    if (d < D) {
      float v = acc[i] + bias[h * D + d];
      v = v > 0.f ? v : (__expf(v) - 1.f);         // ELU
      orow[d] = f32_to_bf16_bits(v);
    }
  }
}

// ------------------- layer-3 aggregate + head mean -------------------------
// H=3 heads, D=64; one wave per node (lane covers d and d+32); f32 out.
__global__ void __launch_bounds__(256)
gat_aggregate_mean_kernel(const float* __restrict__ Hbuf,
                          const int* __restrict__ src,
                          const float* __restrict__ EL, const float* __restrict__ ER,
                          const float* __restrict__ bias,
                          float* __restrict__ out) {
  const int Hn = 3, D = 64;
  int n    = (blockIdx.x * blockDim.x + threadIdx.x) >> 5;
  int lane = threadIdx.x & 31;
  if (n >= NNODES) return;

  int s[DEG];
#pragma unroll
  for (int k = 0; k < DEG; ++k) s[k] = src[n * DEG + k];

  float o0 = 0.f, o1 = 0.f;
  for (int h = 0; h < Hn; ++h) {
    float er_n = ER[n * Hn + h];
    float e[DEG];
    float m = -3.0e38f;
#pragma unroll
    for (int k = 0; k < DEG; ++k) {
      float ev = EL[s[k] * Hn + h] + er_n;
      ev = ev > 0.f ? ev : 0.2f * ev;
      e[k] = ev;
      m = fmaxf(m, ev);
    }
    float denom = 0.f;
#pragma unroll
    for (int k = 0; k < DEG; ++k) { e[k] = __expf(e[k] - m); denom += e[k]; }
    float inv = 1.f / denom;

    float a0 = 0.f, a1 = 0.f;
    for (int k = 0; k < DEG; ++k) {
      const float* hr = Hbuf + (size_t)s[k] * (Hn * D) + h * D;
      float w = e[k] * inv;
      a0 = fmaf(w, hr[lane], a0);
      a1 = fmaf(w, hr[lane + 32], a1);
    }
    o0 += a0 + bias[h * D + lane];
    o1 += a1 + bias[h * D + lane + 32];
  }
  out[(size_t)n * D + lane]      = o0 * (1.f / 3.f);
  out[(size_t)n * D + lane + 32] = o1 * (1.f / 3.f);
}

// ------------------------------- driver ------------------------------------

extern "C" void kernel_launch(void* const* d_in, const int* in_sizes, int n_in,
                              void* d_out, int out_size, void* d_ws, size_t ws_size,
                              hipStream_t stream) {
  (void)in_sizes; (void)n_in; (void)out_size; (void)ws_size;

  const float* nfeat = (const float*)d_in[0];
  const int*   src   = (const int*)d_in[1];
  // d_in[2] = dst: structurally repeat(arange(N),16) -> segments are implicit.
  const float* W1  = (const float*)d_in[3];
  const float* al1 = (const float*)d_in[4];
  const float* ar1 = (const float*)d_in[5];
  const float* b1  = (const float*)d_in[6];
  const float* W2  = (const float*)d_in[7];
  const float* al2 = (const float*)d_in[8];
  const float* ar2 = (const float*)d_in[9];
  const float* b2  = (const float*)d_in[10];
  const float* W3  = (const float*)d_in[11];
  const float* al3 = (const float*)d_in[12];
  const float* ar3 = (const float*)d_in[13];
  const float* b3  = (const float*)d_in[14];

  // workspace carve (max widths: Fin/Fout <= 640)
  char* ws = (char*)d_ws;
  size_t off = 0;
  unsigned short* Xa   = (unsigned short*)(ws + off); off += (size_t)NNODES * 640 * 2;
  unsigned short* Xb   = (unsigned short*)(ws + off); off += (size_t)NNODES * 640 * 2;
  float*          Hbuf = (float*)(ws + off);          off += (size_t)NNODES * 640 * 4;
  unsigned short* Wt   = (unsigned short*)(ws + off); off += (size_t)640 * 640 * 2;
  float*          EL   = (float*)(ws + off);          off += (size_t)NNODES * 8 * 4;
  float*          ER   = (float*)(ws + off);          off += (size_t)NNODES * 8 * 4;

  dim3 blk(256);

  // ---- layer 1: 64 -> 3 heads x 128 ----
  cvt_bf16_kernel<<<1024, blk, 0, stream>>>(nfeat, Xa, NNODES * 64);
  transpose_bf16_kernel<<<(64 * 384 + 255) / 256, blk, 0, stream>>>(W1, Wt, 64, 384);
  gemm_bf16_wmma_kernel<<<dim3(NNODES / 128, 384 / 64), blk, 0, stream>>>(Xa, Wt, Hbuf, 64, 384);
  attn_scores_kernel<<<NNODES * 3 / 8, blk, 0, stream>>>(Hbuf, al1, ar1, EL, ER, 3, 128);
  gat_aggregate_elu_kernel<<<NNODES * 3 / 8, blk, 0, stream>>>(Hbuf, src, EL, ER, b1, Xb, 3, 128);

  // ---- layer 2: 384 -> 5 heads x 128 ----
  transpose_bf16_kernel<<<(384 * 640 + 255) / 256, blk, 0, stream>>>(W2, Wt, 384, 640);
  gemm_bf16_wmma_kernel<<<dim3(NNODES / 128, 640 / 64), blk, 0, stream>>>(Xb, Wt, Hbuf, 384, 640);
  attn_scores_kernel<<<NNODES * 5 / 8, blk, 0, stream>>>(Hbuf, al2, ar2, EL, ER, 5, 128);
  gat_aggregate_elu_kernel<<<NNODES * 5 / 8, blk, 0, stream>>>(Hbuf, src, EL, ER, b2, Xa, 5, 128);

  // ---- layer 3: 640 -> 3 heads x 64, mean over heads ----
  transpose_bf16_kernel<<<(640 * 192 + 255) / 256, blk, 0, stream>>>(W3, Wt, 640, 192);
  gemm_bf16_wmma_kernel<<<dim3(NNODES / 128, 192 / 64), blk, 0, stream>>>(Xa, Wt, Hbuf, 640, 192);
  attn_scores_kernel<<<NNODES * 3 / 8, blk, 0, stream>>>(Hbuf, al3, ar3, EL, ER, 3, 64);
  gat_aggregate_mean_kernel<<<NNODES / 8, blk, 0, stream>>>(Hbuf, src, EL, ER, b3, (float*)d_out);
}
